// AutoEncoder_40312563040596
// MI455X (gfx1250) — compile-verified, matrix operand order checked by reading, not run on hardware
//
#include <hip/hip_runtime.h>
#include <hip/hip_bf16.h>
#include <cstdint>

// ---------------------------------------------------------------------------
// CDNA5 (gfx1250) fused autoencoder: bf16 WMMA GEMM + LayerNorm + Mish layers
// M=32 row tiles; software-pipelined K loop; TDM (tensor_load_to_lds) stages
// the A tile into LDS via the Tensor Data Mover.
// ---------------------------------------------------------------------------

typedef __attribute__((ext_vector_type(16))) __bf16 v16bf;
typedef __attribute__((ext_vector_type(8)))  __bf16 v8bf;
typedef __attribute__((ext_vector_type(8)))  float  v8f;
typedef __attribute__((ext_vector_type(4)))  unsigned int u32x4;
typedef __attribute__((ext_vector_type(8)))  int i32x8;
typedef __attribute__((ext_vector_type(4)))  int i32x4;

union ABFrag { v16bf v; v8bf h[2]; };

struct Grp { ABFrag a0, a1, b; };

__device__ __forceinline__ void load_grp(Grp& g,
                                         const __bf16* arow0,
                                         const __bf16* arow1,
                                         const __bf16* wrow, int k)
{
    g.a0.h[0] = *(const v8bf*)(arow0 + k);
    g.a0.h[1] = *(const v8bf*)(arow0 + k + 16);
    g.a1.h[0] = *(const v8bf*)(arow1 + k);
    g.a1.h[1] = *(const v8bf*)(arow1 + k + 16);
    g.b.h[0]  = *(const v8bf*)(wrow + k);
    g.b.h[1]  = *(const v8bf*)(wrow + k + 16);
}

__device__ __forceinline__ float mishf(float x) {
    float sp = log1pf(expf(x));      // softplus
    return x * tanhf(sp);
}

// ---------------------------------------------------------------------------
// TDM: DMA a (rows x Kp) bf16 row-major tile from global memory into LDS
// offset 0.  All descriptor fields are wave-uniform (SGPRs).  Issued by one
// wave; TENSORcnt tracks completion.  (6-arg clang-23 builtin variant.)
// ---------------------------------------------------------------------------
__device__ __forceinline__ void tdm_stage_tile(const __bf16* gsrc, int Kp, int rows)
{
#if __has_builtin(__builtin_amdgcn_tensor_load_to_lds) && __has_builtin(__builtin_amdgcn_s_wait_tensorcnt)
    if (threadIdx.x < 32) {
        unsigned long long ga = (unsigned long long)(uintptr_t)gsrc;
        u32x4 g0;
        g0[0] = 1u;                                              // count=1
        g0[1] = 0u;                                              // lds_addr = 0
        g0[2] = (unsigned)(ga & 0xFFFFFFFFu);                    // global_addr[31:0]
        g0[3] = (unsigned)((ga >> 32) & 0x01FFFFFFu)             // global_addr[56:32]
              | 0x80000000u;                                     // type=2 ("image")
        i32x8 g1;
        g1[0] = 1 << 16;                                         // data_size=2B
        g1[1] = (Kp & 0xFFFF) << 16;                             // tensor_dim0 lo16
        g1[2] = ((Kp >> 16) & 0xFFFF) | ((rows & 0xFFFF) << 16); // dim0 hi | dim1 lo
        g1[3] = ((rows >> 16) & 0xFFFF) | ((Kp & 0xFFFF) << 16); // dim1 hi | tile_dim0
        g1[4] = rows & 0xFFFF;                                   // tile_dim1 (tile_dim2=0)
        g1[5] = Kp;                                              // tensor_dim0_stride lo32
        g1[6] = 0;                                               // stride hi | dim1_stride lo
        g1[7] = 0;
        i32x4 z4 = {0, 0, 0, 0};
        i32x8 z8 = {0, 0, 0, 0, 0, 0, 0, 0};
        __builtin_amdgcn_tensor_load_to_lds(g0, g1, z4, z4, z8, 0);
        __builtin_amdgcn_s_wait_tensorcnt(0);
    }
#else
    {
        const int4* src = (const int4*)gsrc;
        extern __shared__ char smem_fb[];
        int4* dst = (int4*)smem_fb;
        int nchunks = (rows * Kp * 2) >> 4;
        for (int i = threadIdx.x; i < nchunks; i += 256) dst[i] = src[i];
    }
#endif
}

// ---------------------------------------------------------------------------
// fp32 -> bf16 with zero padding to (Rp x Cp).  Used for x and all weights.
// ---------------------------------------------------------------------------
__global__ __launch_bounds__(256) void convert_pad_bf16(
    const float* __restrict__ src, __bf16* __restrict__ dst,
    int R, int C, int Rp, int Cp, long long total)
{
    long long idx = (long long)blockIdx.x * 256 + threadIdx.x;
    if (idx >= total) return;
    int r = (int)(idx / Cp);
    int c = (int)(idx % Cp);
    float v = (r < R && c < C) ? src[(size_t)r * C + c] : 0.0f;
    dst[idx] = (__bf16)v;
}

__global__ void zero_scalar(float* p) { *p = 0.0f; }

// ---------------------------------------------------------------------------
// Core GEMM body: 32 batch rows x Np cols per block, 8 waves.
// Each wave owns a 32x16 output slab per column pass (two 16x16 WMMA tiles
// sharing one B fragment).  Kp must be a multiple of 64 and >= 64.
// ---------------------------------------------------------------------------
__device__ __forceinline__ void gemm_body(
    const __bf16* __restrict__ A, const __bf16* __restrict__ Wb,
    __bf16* Alds, float* Olds, int Kp, int Np, int rowBase)
{
    const int tid  = threadIdx.x;
    const int lane = tid & 31;
    const int w    = tid >> 5;          // wave 0..7
    const int m    = lane & 15;         // M row / N col within tile
    const int kh   = lane >> 4;         // K-half select per ISA layout

    // stage A tile (32 rows x Kp bf16) into LDS via the Tensor Data Mover
    tdm_stage_tile(A + (size_t)rowBase * Kp, Kp, 32);
    __syncthreads();

    const __bf16* arow0 = Alds + (size_t)m * Kp + kh * 8;         // rows 0..15
    const __bf16* arow1 = Alds + (size_t)(m + 16) * Kp + kh * 8;  // rows 16..31

    for (int c0 = 0; c0 < Np; c0 += 128) {
        int c = c0 + w * 16;
        if (c < Np) {                   // wave-uniform: EXEC stays all-1s
            v8f acc0 = {}, acc1 = {};
            const __bf16* wrow = Wb + (size_t)(c + m) * Kp + kh * 8;

            Grp g0, g1;
            load_grp(g0, arow0, arow1, wrow, 0);
            load_grp(g1, arow0, arow1, wrow, 32);

            int k0 = 0;
            for (; k0 < Kp - 64; k0 += 64) {
                acc0 = __builtin_amdgcn_wmma_f32_16x16x32_bf16(
                        false, g0.a0.v, false, g0.b.v, (short)0, acc0, false, false);
                acc1 = __builtin_amdgcn_wmma_f32_16x16x32_bf16(
                        false, g0.a1.v, false, g0.b.v, (short)0, acc1, false, false);
                load_grp(g0, arow0, arow1, wrow, k0 + 64);
                __builtin_prefetch((const void*)(wrow + k0 + 128), 0, 3);
                acc0 = __builtin_amdgcn_wmma_f32_16x16x32_bf16(
                        false, g1.a0.v, false, g1.b.v, (short)0, acc0, false, false);
                acc1 = __builtin_amdgcn_wmma_f32_16x16x32_bf16(
                        false, g1.a1.v, false, g1.b.v, (short)0, acc1, false, false);
                load_grp(g1, arow0, arow1, wrow, k0 + 96);
            }
            // tail: consume the last two primed groups
            acc0 = __builtin_amdgcn_wmma_f32_16x16x32_bf16(
                    false, g0.a0.v, false, g0.b.v, (short)0, acc0, false, false);
            acc1 = __builtin_amdgcn_wmma_f32_16x16x32_bf16(
                    false, g0.a1.v, false, g0.b.v, (short)0, acc1, false, false);
            acc0 = __builtin_amdgcn_wmma_f32_16x16x32_bf16(
                    false, g1.a0.v, false, g1.b.v, (short)0, acc0, false, false);
            acc1 = __builtin_amdgcn_wmma_f32_16x16x32_bf16(
                    false, g1.a1.v, false, g1.b.v, (short)0, acc1, false, false);

            // C layout: VGPR r -> (M = r + 8*kh, N = lane&15)
            #pragma unroll
            for (int r = 0; r < 8; ++r) {
                Olds[(size_t)(r + kh * 8) * Np + c + m]      = acc0[r];
                Olds[(size_t)(r + kh * 8 + 16) * Np + c + m] = acc1[r];
            }
        }
    }
    __syncthreads();
}

// ---------------------------------------------------------------------------
// Fused  out = Mish(LayerNorm(A @ W^T (+bias)))  for one layer.
//   A   : B x Kp   bf16 (zero padded K)
//   Wb  : Np x Kp  bf16 (zero padded rows/cols)
//   out : B x Np   bf16 (padding columns written as 0)
// ---------------------------------------------------------------------------
__global__ __launch_bounds__(256) void gemm_ln_mish(
    const __bf16* __restrict__ A, const __bf16* __restrict__ Wb,
    const float* __restrict__ bias,          // may be nullptr (pre-LN bias)
    const float* __restrict__ gamma, const float* __restrict__ beta,
    __bf16* __restrict__ outA,
    int Kp, int N, int Np,
    float* __restrict__ latSum)              // non-null for latent layer
{
    extern __shared__ char smem[];
    __bf16* Alds = (__bf16*)smem;                               // 32*Kp bf16 (at LDS 0)
    float*  Olds = (float*)(smem + (size_t)32 * Kp * 2);        // 32*Np f32
    float*  redS = Olds + (size_t)32 * Np;                      // 256 f32
    float*  redQ = redS + 256;                                  // 256 f32

    const int tid = threadIdx.x;
    const int rowBase = blockIdx.x * 32;

    gemm_body(A, Wb, Alds, Olds, Kp, Np, rowBase);

    // ---- LayerNorm + Mish epilogue: 8 threads per row, 32 rows ----
    const int r = tid >> 3;             // row 0..31
    const int l = tid & 7;              // lane-in-row 0..7

    float s = 0.0f, q = 0.0f;
    for (int j = l; j < N; j += 8) {
        float v = Olds[(size_t)r * Np + j];
        if (bias) v += bias[j];
        s += v; q += v * v;
    }
    redS[r * 8 + l] = s;
    redQ[r * 8 + l] = q;
    __syncthreads();
    if (l == 0) {
        float S = 0.0f, Q = 0.0f;
        #pragma unroll
        for (int t = 0; t < 8; ++t) { S += redS[r * 8 + t]; Q += redQ[r * 8 + t]; }
        float mu  = S / (float)N;
        float var = Q / (float)N - mu * mu;
        redS[r * 8] = mu;
        redQ[r * 8] = rsqrtf(var + 1e-5f);
    }
    __syncthreads();
    const float mu   = redS[r * 8];
    const float rstd = redQ[r * 8];

    float lsum = 0.0f;
    for (int j = l; j < Np; j += 8) {
        float v = 0.0f;
        if (j < N) {
            float h = Olds[(size_t)r * Np + j];
            if (bias) h += bias[j];
            h = (h - mu) * rstd * gamma[j] + beta[j];
            v = mishf(h);
            lsum += fabsf(v);
        }
        outA[(size_t)(rowBase + r) * Np + j] = (__bf16)v;
    }

    if (latSum) {
        __syncthreads();
        redS[tid] = lsum;
        __syncthreads();
        if (tid == 0) {
            float t = 0.0f;
            for (int i = 0; i < 256; ++i) t += redS[i];
            atomicAdd(latSum, t);
        }
    }
}

// ---------------------------------------------------------------------------
// Final layer:  out = sigmoid(A @ W^T + bo),  fp32 output, unpadded width N.
// ---------------------------------------------------------------------------
__global__ __launch_bounds__(256) void gemm_sigmoid(
    const __bf16* __restrict__ A, const __bf16* __restrict__ Wb,
    const float* __restrict__ bo, float* __restrict__ out,
    int Kp, int N, int Np)
{
    extern __shared__ char smem[];
    __bf16* Alds = (__bf16*)smem;
    float*  Olds = (float*)(smem + (size_t)32 * Kp * 2);

    const int tid = threadIdx.x;
    const int rowBase = blockIdx.x * 32;

    gemm_body(A, Wb, Alds, Olds, Kp, Np, rowBase);

    const int r = tid >> 3;
    const int l = tid & 7;
    for (int j = l; j < N; j += 8) {
        float v = Olds[(size_t)r * Np + j] + bo[j];
        out[(size_t)(rowBase + r) * N + j] = 1.0f / (1.0f + expf(-v));
    }
}

// ---------------------------------------------------------------------------
// Host launcher
// ---------------------------------------------------------------------------
extern "C" void kernel_launch(void* const* d_in, const int* in_sizes, int n_in,
                              void* d_out, int out_size, void* d_ws, size_t ws_size,
                              hipStream_t stream) {
    const int B = 65536;
    const int D = 784,  Dp = 832;     // pad to multiple of 64 (K dims)
    const int O1 = 512, O1p = 512;
    const int O2 = 264, O2p = 320;
    const int O3 = 128, O3p = 128;

    const float* x   = (const float*)d_in[0];
    const float* W1  = (const float*)d_in[1];
    const float* g1  = (const float*)d_in[2];
    const float* b1  = (const float*)d_in[3];
    const float* W2  = (const float*)d_in[4];
    const float* g2  = (const float*)d_in[5];
    const float* b2  = (const float*)d_in[6];
    const float* W3  = (const float*)d_in[7];
    const float* b3  = (const float*)d_in[8];
    const float* gh  = (const float*)d_in[9];
    const float* bh  = (const float*)d_in[10];
    const float* D1  = (const float*)d_in[11];
    const float* g3  = (const float*)d_in[12];
    const float* b3n = (const float*)d_in[13];
    const float* D2  = (const float*)d_in[14];
    const float* g4  = (const float*)d_in[15];
    const float* b4  = (const float*)d_in[16];
    const float* D3  = (const float*)d_in[17];
    const float* bo  = (const float*)d_in[18];

    float* out    = (float*)d_out;
    float* latSum = out + (size_t)B * D;   // scalar slot (tuple second output)

    // ---- carve workspace ----
    char* ws = (char*)d_ws;
    auto carve = [&](size_t bytes) {
        char* p = ws; ws += (bytes + 255) & ~(size_t)255; return p;
    };
    __bf16* xb  = (__bf16*)carve((size_t)B * Dp  * 2);
    __bf16* h1  = (__bf16*)carve((size_t)B * O1p * 2);
    __bf16* h2  = (__bf16*)carve((size_t)B * O2p * 2);
    __bf16* lat = (__bf16*)carve((size_t)B * O3p * 2);
    __bf16* h4  = (__bf16*)carve((size_t)B * O2p * 2);
    __bf16* h5  = (__bf16*)carve((size_t)B * O1p * 2);
    __bf16* Wb1 = (__bf16*)carve((size_t)O1p * Dp  * 2);
    __bf16* Wb2 = (__bf16*)carve((size_t)O2p * O1p * 2);
    __bf16* Wb3 = (__bf16*)carve((size_t)O3p * O2p * 2);
    __bf16* Db1 = (__bf16*)carve((size_t)O2p * O3p * 2);
    __bf16* Db2 = (__bf16*)carve((size_t)O1p * O2p * 2);
    __bf16* Db3 = (__bf16*)carve((size_t)Dp  * O1p * 2);

    auto cvt = [&](const float* src, __bf16* dst, int R, int C, int Rp, int Cp) {
        long long total = (long long)Rp * Cp;
        int blocks = (int)((total + 255) / 256);
        convert_pad_bf16<<<blocks, 256, 0, stream>>>(src, dst, R, C, Rp, Cp, total);
    };

    // ---- pad/convert inputs + weights to bf16 ----
    cvt(x,  xb,  B,   D,   B,   Dp);
    cvt(W1, Wb1, O1,  D,   O1p, Dp);
    cvt(W2, Wb2, O2,  O1,  O2p, O1p);
    cvt(W3, Wb3, O3,  O2,  O3p, O2p);
    cvt(D1, Db1, O2,  O3,  O2p, O3p);
    cvt(D2, Db2, O1,  O2,  O1p, O2p);
    cvt(D3, Db3, D,   O1,  Dp,  O1p);

    zero_scalar<<<1, 1, 0, stream>>>(latSum);

    const int nblk = B / 32;   // 2048 row-tiles
    auto smem = [](int Kp, int Np) {
        return (size_t)32 * Kp * 2 + (size_t)32 * Np * 4 + 2 * 256 * 4;
    };

    // encoder
    gemm_ln_mish<<<nblk, 256, smem(Dp,  O1p), stream>>>(
        xb,  Wb1, nullptr, g1, b1, h1,  Dp,  O1, O1p, nullptr);
    gemm_ln_mish<<<nblk, 256, smem(O1p, O2p), stream>>>(
        h1,  Wb2, nullptr, g2, b2, h2,  O1p, O2, O2p, nullptr);
    gemm_ln_mish<<<nblk, 256, smem(O2p, O3p), stream>>>(
        h2,  Wb3, b3,      gh, bh, lat, O2p, O3, O3p, latSum);
    // decoder
    gemm_ln_mish<<<nblk, 256, smem(O3p, O2p), stream>>>(
        lat, Db1, nullptr, g3, b3n, h4, O3p, O2, O2p, nullptr);
    gemm_ln_mish<<<nblk, 256, smem(O2p, O1p), stream>>>(
        h4,  Db2, nullptr, g4, b4,  h5, O2p, O1, O1p, nullptr);
    gemm_sigmoid<<<nblk, 256, smem(O1p, Dp), stream>>>(
        h5,  Db3, bo, out, O1p, D, Dp);
}
